// CrossWinAttention_11819749998927
// MI455X (gfx1250) — matrix operand order repaired and checked by reading, not compiled, not run on hardware
//
#include <hip/hip_runtime.h>
#include <stdint.h>

typedef __attribute__((ext_vector_type(16))) _Float16 v16h;
typedef __attribute__((ext_vector_type(8)))  _Float16 v8h;
typedef __attribute__((ext_vector_type(8)))  float    v8f;

#define B_  2
#define N_  4
#define X_  16
#define Y_  16
#define W1_ 8
#define W2_ 8
#define D_  128
#define M_  4
#define DH_ 32
#define L_  256
#define T_  256
#define LT_ (L_*T_)    /* 65536 */
#define TD_ (T_*DH_)   /* 8192  */

__device__ __forceinline__ v8f wmma_f16(v16h a, v16h b, v8f c) {
  return __builtin_amdgcn_wmma_f32_16x16x32_f16(false, a, false, b, (short)0, c, false, false);
}
// A-fragment per lane = two contiguous 8xf16 runs, 16 elements apart (K split by lane half)
__device__ __forceinline__ v16h load_frag_split(const _Float16* p) {
  v8h lo = *(const v8h*)p;
  v8h hi = *(const v8h*)(p + 16);
  return __builtin_shufflevector(lo, hi, 0,1,2,3,4,5,6,7,8,9,10,11,12,13,14,15);
}

// --------------------------------------------- weights -> f16, WMMA-B fragment order
// frag(kt,ct): ws[((kt*8+ct)*32 + lane)*16 + e] = W[(kt*32 + (lane>>4)*8 + e + (e>>3)*8)*128 + ct*16 + (lane&15)]
__global__ void convert_w_kernel(const float* __restrict__ wq, const float* __restrict__ wk,
                                 const float* __restrict__ wv, const float* __restrict__ wp,
                                 _Float16* __restrict__ out) {
  int idx  = blockIdx.x * blockDim.x + threadIdx.x;   // 4 mats * 4 kt * 8 ct * 32 lanes = 4096
  int lane = idx & 31;
  int ct   = (idx >> 5) & 7;
  int kt   = (idx >> 8) & 3;
  int mat  = idx >> 10;
  const float* s = (mat == 0) ? wq : (mat == 1) ? wk : (mat == 2) ? wv : wp;
  _Float16* o = out + (size_t)idx * 16;
  int n = lane & 15, kb = (lane >> 4) << 3;
  #pragma unroll
  for (int e = 0; e < 16; ++e)
    o[e] = (_Float16)s[(size_t)(kt*32 + kb + e + ((e >> 3) << 3))*128 + ct*16 + n];
}

// ------------------------------------------------- LN + input projection (GEMM)
// src (B,N,X,Y,W1,W2,D) f32 -> dst f16: transposed==0 -> (B,M,L,T,Dh); ==1 -> (B,M,L,Dh,T)
__global__ __launch_bounds__(128, 1) void proj_in_kernel(
    const float* __restrict__ src, const float* __restrict__ gam, const float* __restrict__ bet,
    const _Float16* __restrict__ wfr, const float* __restrict__ bias,
    _Float16* __restrict__ dst, int transposed)
{
  __shared__ alignas(32) _Float16 an[16*136];
  __shared__ alignas(32) _Float16 outt[16*128];
  int tid  = threadIdx.x;
  int tok0 = blockIdx.x << 4;
  int b  = tok0 >> 16;          // L*T = 65536
  int l  = (tok0 >> 8) & 255;   // T = 256
  int t0 = tok0 & 255;
  { // stage 16 rows with fused LayerNorm (8 lanes per row, same wave)
    int row = tid >> 3, col0 = (tid & 7) << 4;
    int t = t0 + row;
    int n = t >> 6, w1 = (t >> 3) & 7, w2 = t & 7;
    int x = l >> 4,  y = l & 15;
    const float4* rp = (const float4*)(src +
        (size_t)(((((b*N_ + n)*X_ + x)*Y_ + y)*W1_ + w1)*W2_ + w2) * D_ + col0);
    const float4* g4 = (const float4*)(gam + col0);
    const float4* b4 = (const float4*)(bet + col0);
    float va[16]; float s1 = 0.f, s2 = 0.f;
    #pragma unroll
    for (int i = 0; i < 4; ++i) {
      float4 f = rp[i];
      va[4*i+0]=f.x; va[4*i+1]=f.y; va[4*i+2]=f.z; va[4*i+3]=f.w;
    }
    #pragma unroll
    for (int e = 0; e < 16; ++e) { s1 += va[e]; s2 += va[e]*va[e]; }
    #pragma unroll
    for (int o = 1; o < 8; o <<= 1) { s1 += __shfl_xor(s1, o, 8); s2 += __shfl_xor(s2, o, 8); }
    float mu = s1 * (1.f/128.f);
    float rs = rsqrtf(s2 * (1.f/128.f) - mu*mu + 1e-5f);
    #pragma unroll
    for (int i = 0; i < 4; ++i) {
      float4 g = g4[i], bb = b4[i];
      an[row*136 + col0 + 4*i+0] = (_Float16)((va[4*i+0]-mu)*rs*g.x + bb.x);
      an[row*136 + col0 + 4*i+1] = (_Float16)((va[4*i+1]-mu)*rs*g.y + bb.y);
      an[row*136 + col0 + 4*i+2] = (_Float16)((va[4*i+2]-mu)*rs*g.z + bb.z);
      an[row*136 + col0 + 4*i+3] = (_Float16)((va[4*i+3]-mu)*rs*g.w + bb.w);
    }
  }
  __syncthreads();
  int wave = tid >> 5, lane = tid & 31;
  int arow = lane & 15, kbA = (lane >> 4) << 3;
  int ncol = lane & 15, mhi = (lane >> 4) << 3;
  v16h a[4];
  #pragma unroll
  for (int kt = 0; kt < 4; ++kt)
    a[kt] = load_frag_split(an + arow*136 + kt*32 + kbA);
  #pragma unroll
  for (int c2 = 0; c2 < 2; ++c2) {
    int ct = wave*2 + c2;
    v8f acc = {};
    #pragma unroll
    for (int kt = 0; kt < 4; ++kt) {
      v16h bf = *(const v16h*)(wfr + ((size_t)((kt*8 + ct)*32 + lane))*16);
      acc = wmma_f16(a[kt], bf, acc);
    }
    int ch = ct*16 + ncol;
    float bs = bias[ch];
    if (!transposed) {
      #pragma unroll
      for (int g = 0; g < 8; ++g) outt[(mhi+g)*128 + ch] = (_Float16)(acc[g] + bs);
    } else {
      #pragma unroll
      for (int g = 0; g < 8; ++g) outt[ch*16 + mhi + g] = (_Float16)(acc[g] + bs);
    }
  }
  __syncthreads();
  if (!transposed) {        // dst[((b*M+m)*LT + l*T + t)*32 + dh], vector 16B chunks
    #pragma unroll
    for (int it = 0; it < 2; ++it) {
      int ck = it*128 + tid;          // 256 chunks of 8 f16
      int trow = ck >> 4, co = (ck & 15) << 3;
      int m = co >> 5, dh = co & 31;
      v8h d = *(const v8h*)(outt + trow*128 + co);
      *(v8h*)(dst + (size_t)((b*M_+m)*LT_ + l*T_ + t0 + trow)*DH_ + dh) = d;
    }
  } else {                  // dst[((b*M+m)*L + l)*Dh*T + dh*T + t], one 32B row per thread
    int ch = tid, m = ch >> 5, dh = ch & 31;
    v16h d = *(const v16h*)(outt + ch*16);
    *(v16h*)(dst + ((size_t)(b*M_+m)*L_ + l)*(DH_*T_) + (size_t)dh*T_ + t0) = d;
  }
}

// ------------------------------------------------------------------- attention
// block = (b,m,l) x 64-query tile; K/V^T async-staged to LDS; wave owns 16 queries
__global__ __launch_bounds__(128, 1) void attn_kernel(
    const _Float16* __restrict__ qh, const _Float16* __restrict__ kh,
    const _Float16* __restrict__ vhT, const float* __restrict__ temp,
    _Float16* __restrict__ ab)
{
  __shared__ alignas(32) _Float16 Ksh[256*32];     // [key][dh]  16KB (reused as out staging)
  __shared__ alignas(32) _Float16 Vsh[32*256];     // [dh][key]  16KB
  __shared__ alignas(32) _Float16 att[4*16*256];   // per-wave probabilities 32KB
  int tid  = threadIdx.x;
  int wave = tid >> 5, lane = tid & 31;
  int bml  = blockIdx.x >> 2;
  int qt   = blockIdx.x & 3;
  int b    = bml >> 10;                  // M*L = 1024
  size_t base = (size_t)bml * TD_;
  int q0   = qt*64 + wave*16;
  float scale = 0.17677669529663687f / temp[b];   // 32^-0.5 / temperature[b]
  int arow = lane & 15, kbA = (lane >> 4) << 3;
  int ncol = lane & 15, kbB = (lane >> 4) << 4, mhi = (lane >> 4) << 3;

  { // async-stage K (16KB) and V^T (16KB) into LDS: 128 lanes x 8 x 16B each
    unsigned ldsK = (unsigned)(size_t)(void*)Ksh;
    unsigned ldsV = (unsigned)(size_t)(void*)Vsh;
    unsigned long long gK = (unsigned long long)(uintptr_t)(kh  + base);
    unsigned long long gV = (unsigned long long)(uintptr_t)(vhT + base);
    #pragma unroll
    for (int it = 0; it < 8; ++it) {
      unsigned off = (unsigned)((it*128 + tid) * 16);
      unsigned lk = ldsK + off, lv = ldsV + off;
      unsigned long long ak = gK + off, av = gV + off;
      asm volatile("global_load_async_to_lds_b128 %0, %1, off" :: "v"(lk), "v"(ak) : "memory");
      asm volatile("global_load_async_to_lds_b128 %0, %1, off" :: "v"(lv), "v"(av) : "memory");
    }
    asm volatile("s_wait_asynccnt 0x0" ::: "memory");
  }
  __syncthreads();

  v16h aq = load_frag_split(qh + base + (size_t)(q0 + arow)*32 + kbA);  // K = Dh = 32
  v8f c[16];                             // 16 score tiles = all 256 keys
  #pragma unroll
  for (int ct = 0; ct < 16; ++ct) {
    v16h bf = load_frag_split(Ksh + (ct*16 + ncol)*32 + kbA);
    v8f z = {};
    c[ct] = wmma_f16(aq, bf, z);
  }
  // register softmax: each score row lives in one 16-lane half
  float mx[8], sm[8];
  #pragma unroll
  for (int g = 0; g < 8; ++g) mx[g] = -3.0e38f;
  #pragma unroll
  for (int ct = 0; ct < 16; ++ct)
    #pragma unroll
    for (int g = 0; g < 8; ++g) { float v = c[ct][g]*scale; c[ct][g] = v; mx[g] = fmaxf(mx[g], v); }
  #pragma unroll
  for (int g = 0; g < 8; ++g)
    #pragma unroll
    for (int o = 1; o < 16; o <<= 1) mx[g] = fmaxf(mx[g], __shfl_xor(mx[g], o, 32));
  #pragma unroll
  for (int g = 0; g < 8; ++g) sm[g] = 0.f;
  #pragma unroll
  for (int ct = 0; ct < 16; ++ct)
    #pragma unroll
    for (int g = 0; g < 8; ++g) { float e = __expf(c[ct][g] - mx[g]); c[ct][g] = e; sm[g] += e; }
  #pragma unroll
  for (int g = 0; g < 8; ++g) {
    #pragma unroll
    for (int o = 1; o < 16; o <<= 1) sm[g] += __shfl_xor(sm[g], o, 32);
    sm[g] = 1.f / sm[g];
  }
  _Float16* attw = att + wave*4096;
  #pragma unroll
  for (int ct = 0; ct < 16; ++ct)
    #pragma unroll
    for (int g = 0; g < 8; ++g)
      attw[(mhi + g)*256 + ct*16 + ncol] = (_Float16)(c[ct][g] * sm[g]);
  __syncthreads();                       // also retires all Ksh readers (allows overlay)

  _Float16* obuf = Ksh;                  // out staging overlays dead K tile
  #pragma unroll
  for (int vt = 0; vt < 2; ++vt) {       // P @ V: 8 accumulating WMMAs per 16x16 tile
    v8f acc = {};
    #pragma unroll
    for (int kc = 0; kc < 8; ++kc) {
      v16h af = load_frag_split(attw + arow*256 + kc*32 + kbA);
      v16h bf = load_frag_split(Vsh + (vt*16 + ncol)*256 + kc*32 + kbA);
      acc = wmma_f16(af, bf, acc);
    }
    #pragma unroll
    for (int g = 0; g < 8; ++g)
      obuf[wave*512 + (mhi+g)*32 + vt*16 + ncol] = (_Float16)acc[g];
  }
  __syncthreads();
  #pragma unroll
  for (int rep = 0; rep < 2; ++rep) {    // vectorized writeback: 16B per lane per rep
    int ckid = rep*32 + lane;
    int r = ckid >> 2, h = (ckid & 3) << 3;
    v8h d = *(const v8h*)(obuf + wave*512 + r*32 + h);
    *(v8h*)(ab + base + (size_t)(q0 + r)*32 + h) = d;
  }
}

// ----------------------------- output projection + mean-over-views + skip (fused)
__global__ __launch_bounds__(128, 1) void outproj_kernel(
    const _Float16* __restrict__ ab, const _Float16* __restrict__ wpfr,
    const float* __restrict__ bproj, const float* __restrict__ skip,
    float* __restrict__ out)
{
  __shared__ alignas(32) float outt[16*128];
  int tid  = threadIdx.x;
  int wave = tid >> 5, lane = tid & 31;
  int bl = blockIdx.x >> 2;
  int jt = blockIdx.x & 3;
  int b = bl >> 8, l = bl & 255;
  int j0 = jt << 4;
  int arow = lane & 15, kbA = (lane >> 4) << 3;
  int ncol = lane & 15, mhi = (lane >> 4) << 3;
  v16h afr[16];                          // 16 K-tiles: (view n, head m)
  #pragma unroll
  for (int n = 0; n < 4; ++n)
    #pragma unroll
    for (int m = 0; m < 4; ++m) {
      size_t rb = ((size_t)((b*M_ + m)*L_ + l)*T_ + (n*64 + j0 + arow)) * DH_;
      afr[n*4 + m] = load_frag_split(ab + rb + kbA);
    }
  #pragma unroll
  for (int c2 = 0; c2 < 2; ++c2) {
    int ct = wave*2 + c2;
    v8f acc = {};
    #pragma unroll
    for (int m = 0; m < 4; ++m) {
      v16h bf = *(const v16h*)(wpfr + ((size_t)((m*8 + ct)*32 + lane))*16);
      #pragma unroll
      for (int n = 0; n < 4; ++n)
        acc = wmma_f16(afr[n*4 + m], bf, acc);
    }
    #pragma unroll
    for (int g = 0; g < 8; ++g)
      outt[(mhi+g)*128 + ct*16 + ncol] = acc[g];
  }
  __syncthreads();
  #pragma unroll
  for (int it = 0; it < 4; ++it) {       // 512 chunks of 4 f32; skip/out fully vectorized
    int ck = it*128 + tid;
    int row = ck >> 5, co = (ck & 31) << 2;
    size_t o = ((size_t)(b*L_ + l)*64 + j0 + row)*128 + co;
    float4 a4 = *(const float4*)(outt + row*128 + co);
    float4 sk = *(const float4*)(skip + o);
    float4 bp = *(const float4*)(bproj + co);
    float4 r;
    r.x = a4.x*0.25f + bp.x + sk.x;
    r.y = a4.y*0.25f + bp.y + sk.y;
    r.z = a4.z*0.25f + bp.z + sk.z;
    r.w = a4.w*0.25f + bp.w + sk.w;
    *(float4*)(out + o) = r;
  }
}

extern "C" void kernel_launch(void* const* d_in, const int* in_sizes, int n_in,
                              void* d_out, int out_size, void* d_ws, size_t ws_size,
                              hipStream_t stream) {
  (void)in_sizes; (void)n_in; (void)out_size; (void)ws_size;
  const float* q    = (const float*)d_in[0];
  const float* k    = (const float*)d_in[1];
  const float* v    = (const float*)d_in[2];
  const float* skip = (const float*)d_in[3];
  const float* temp = (const float*)d_in[4];
  const float* lnqg = (const float*)d_in[5];
  const float* lnqb = (const float*)d_in[6];
  const float* lnkg = (const float*)d_in[7];
  const float* lnkb = (const float*)d_in[8];
  const float* lnvg = (const float*)d_in[9];
  const float* lnvb = (const float*)d_in[10];
  const float* wq   = (const float*)d_in[11];
  const float* bq   = (const float*)d_in[12];
  const float* wk   = (const float*)d_in[13];
  const float* bk   = (const float*)d_in[14];
  const float* wv   = (const float*)d_in[15];
  const float* bv   = (const float*)d_in[16];
  const float* wp   = (const float*)d_in[17];
  const float* bp   = (const float*)d_in[18];
  float* out = (float*)d_out;

  const size_t HBUF = (size_t)B_ * M_ * LT_ * DH_;   // 16,777,216 f16 elems
  _Float16* wq16 = (_Float16*)d_ws;                  // fragment-swizzled weights
  _Float16* wk16 = wq16 + 16384;
  _Float16* wv16 = wk16 + 16384;
  _Float16* wp16 = wv16 + 16384;
  _Float16* qh   = wp16 + 16384;                     // (B,M,L,T,Dh)
  _Float16* kh   = qh + HBUF;                        // (B,M,L,T,Dh)
  _Float16* vhT  = kh + HBUF;                        // (B,M,L,Dh,T)
  _Float16* abuf = vhT + HBUF;                       // (B,M,L,T,Dh)

  convert_w_kernel<<<32, 128, 0, stream>>>(wq, wk, wv, wp, wq16);
  proj_in_kernel<<<8192, 128, 0, stream>>>(q, lnqg, lnqb, wq16, bq, qh, 0);
  proj_in_kernel<<<8192, 128, 0, stream>>>(k, lnkg, lnkb, wk16, bk, kh, 0);
  proj_in_kernel<<<8192, 128, 0, stream>>>(v, lnvg, lnvb, wv16, bv, vhT, 1);
  attn_kernel<<<8192, 128, 0, stream>>>(qh, kh, vhT, temp, abuf);
  outproj_kernel<<<2048, 128, 0, stream>>>(abuf, wp16, bp, skip, out);
}
